// _HAttn_10900626997860
// MI455X (gfx1250) — matrix operand order repaired and checked by reading, not compiled, not run on hardware
//
#include <hip/hip_runtime.h>
#include <hip/hip_bf16.h>
#include <math.h>

typedef _Float16 half_t;
typedef __attribute__((ext_vector_type(8)))  _Float16 v8h;
typedef __attribute__((ext_vector_type(16))) _Float16 v16h;
typedef __attribute__((ext_vector_type(8)))  float    v8f;

#define B_    2
#define S_    2048
#define D_    2048
#define H_    16
#define HKV_  4
#define DH_   128
#define REPS_ (H_ / HKV_)
#define M_    (B_ * S_)              // 4096 token rows
#define NQKV_ (D_ + 2 * HKV_ * DH_)  // 3072
#define EPS_  1.1920929e-07f
#define PPITCH 40   // LDS pitch (halves) for P transpose: 20-bank stride
#define WPITCH 72   // LDS pitch (halves) for 64-wide weight slices: 36-bank stride

// ---------------- WMMA fragment helpers (wave32, 16x16x32 f16) ----------------

__device__ __forceinline__ v16h cat16(v8h lo, v8h hi) {
  v16h r;
#pragma unroll
  for (int i = 0; i < 8; ++i) { r[i] = lo[i]; r[i + 8] = hi[i]; }
  return r;
}

// A-matrix 16x32 f16: lane l (0..15) row M=l holds K {0..15};
// lane l+16 row M=l holds K {16..31}.
__device__ __forceinline__ v16h load_a16(const half_t* row, int k0, int hi) {
  const half_t* p = row + k0 + hi * 16;
  return cat16(*reinterpret_cast<const v8h*>(p),
               *reinterpret_cast<const v8h*>(p + 8));
}

// B-matrix 32x16 f16 from global: lane l (=N col); column of B = row of (N,K)
// weight matrix, contiguous.  32B-aligned at all call sites.
__device__ __forceinline__ v16h load_b16(const half_t* colrow, int k0, int hi) {
  return *reinterpret_cast<const v16h*>(colrow + k0 + hi * 16);
}

// B-frag from an LDS weight tile (pitch WPITCH halves, sub-k offset in halves).
__device__ __forceinline__ v16h ldsb_frag(const half_t* buf, int n, int sub, int hi) {
  const half_t* p = buf + n * WPITCH + sub + hi * 16;
  return cat16(*reinterpret_cast<const v8h*>(p),
               *reinterpret_cast<const v8h*>(p + 8));
}

__device__ __forceinline__ v8f wmma16(v16h a, v16h b, v8f c) {
  return __builtin_amdgcn_wmma_f32_16x16x32_f16(false, a, false, b,
                                                (short)0, c, false, false);
}

// ---------------- gfx1250 async global->LDS copy (ASYNCcnt path) ----------------
// Generic pointers to __shared__ carry the LDS byte offset in their low 32 bits
// (see ISA 10.2).

__device__ __forceinline__ void async_g2l_b128(void* lds_dst, const void* gsrc) {
  unsigned int loff = (unsigned int)(unsigned long long)(uintptr_t)lds_dst;
  unsigned long long ga = (unsigned long long)(uintptr_t)gsrc;
  asm volatile("global_load_async_to_lds_b128 %0, %1, off"
               :: "v"(loff), "v"(ga)
               : "memory");
}
__device__ __forceinline__ void wait_async0() {
  asm volatile("s_wait_asynccnt 0x0" ::: "memory");
}

// ---------------- fp32 -> f16 conversion ----------------

__global__ __launch_bounds__(256) void k_f32_to_f16(const float* __restrict__ s,
                                                    half_t* __restrict__ d, int n) {
  int i = blockIdx.x * blockDim.x + threadIdx.x;
  int stride = gridDim.x * blockDim.x;
  for (; i < n; i += stride) d[i] = (half_t)s[i];
}

// ---------------- QKV GEMM + RMSNorm + RoPE + gain epilogue ----------------
// Block: 8 waves share one 128-col head slab of weights.  64-wide k-slices are
// async double-buffered in LDS; each iteration = 16 WMMAs between barriers.
// head columns: 0..15 = Q heads, 16..19 = K heads, 20..23 = V heads.

__global__ __launch_bounds__(256)
void k_qkv(const half_t* __restrict__ x16, const half_t* __restrict__ wqkv,
           const float* __restrict__ q_gain,
           half_t* __restrict__ q16, half_t* __restrict__ k16,
           half_t* __restrict__ vT16) {
  __shared__ __align__(32) half_t ldsB[2 * 128 * WPITCH];

  const int lane = threadIdx.x & 31;
  const int l15 = lane & 15;
  const int hi  = lane >> 4;
  const int wave = threadIdx.x >> 5;
  const int mblock = blockIdx.x / 24;
  const int hc     = blockIdx.x % 24;
  const int m0  = (mblock * 8 + wave) * 16;
  const int n0  = hc * 128;

  v8f z = {};
  v8f c[8];
#pragma unroll
  for (int t = 0; t < 8; ++t) c[t] = z;

  const half_t* arow = x16 + (size_t)(m0 + l15) * D_;

  // cooperative staging: thread -> (row, 32-half chunk) => 4 b128 asyncs each
  const int srow = threadIdx.x >> 1;        // 0..127
  const int sp   = threadIdx.x & 1;         // 0..1  (32-half chunk)
  const half_t* wsrc = wqkv + (size_t)(n0 + srow) * D_ + sp * 32;
  half_t* sdst[2];
  sdst[0] = ldsB + srow * WPITCH + sp * 32;
  sdst[1] = sdst[0] + 128 * WPITCH;

  // prologue: stage k-slice [0,64) into buffer 0
  async_g2l_b128(sdst[0] +  0, wsrc);
  async_g2l_b128(sdst[0] +  8, wsrc + 8);
  async_g2l_b128(sdst[0] + 16, wsrc + 16);
  async_g2l_b128(sdst[0] + 24, wsrc + 24);
  wait_async0();
  __syncthreads();

  v16h a0 = load_a16(arow, 0, hi);
  v16h a1 = load_a16(arow, 32, hi);
  int cur = 0;
  for (int k0 = 0; k0 < D_; k0 += 64) {
    const bool more = (k0 + 64 < D_);
    if (more) {
      const half_t* ns = wsrc + k0 + 64;
      half_t* nd = sdst[cur ^ 1];
      async_g2l_b128(nd +  0, ns);
      async_g2l_b128(nd +  8, ns + 8);
      async_g2l_b128(nd + 16, ns + 16);
      async_g2l_b128(nd + 24, ns + 24);
      __builtin_prefetch(arow + k0 + 128, 0, 1);
    }
    v16h an0 = a0, an1 = a1;
    if (more) { an0 = load_a16(arow, k0 + 64, hi); an1 = load_a16(arow, k0 + 96, hi); }

    const half_t* bb = ldsB + cur * (128 * WPITCH);
    v16h kb[8];
#pragma unroll
    for (int t = 0; t < 8; ++t) kb[t] = ldsb_frag(bb, t * 16 + l15, 0, hi);
#pragma unroll
    for (int t = 0; t < 8; ++t) c[t] = wmma16(a0, kb[t], c[t]);
#pragma unroll
    for (int t = 0; t < 8; ++t) kb[t] = ldsb_frag(bb, t * 16 + l15, 32, hi);
#pragma unroll
    for (int t = 0; t < 8; ++t) c[t] = wmma16(a1, kb[t], c[t]);

    a0 = an0; a1 = an1;
    wait_async0();
    __syncthreads();
    cur ^= 1;
  }

  if (hc < 20) {
    // RMS norm over Dh=128: row m = r + 8*hi lives in 16 lanes of this half.
#pragma unroll
    for (int r = 0; r < 8; ++r) {
      float ss = 0.f;
#pragma unroll
      for (int t = 0; t < 8; ++t) ss += c[t][r] * c[t][r];
#pragma unroll
      for (int off = 1; off < 16; off <<= 1) ss += __shfl_xor(ss, off, 16);
      float sc = rsqrtf(ss * (1.0f / 128.0f) + EPS_);
#pragma unroll
      for (int t = 0; t < 8; ++t) c[t][r] *= sc;
    }
    // RoPE with NTK-scaled base (S=2048 > 1024): base = 1e4 * 2^(128/126)
    const float logbase = logf(10000.0f) + (128.0f / 126.0f) * 0.6931471805599453f;
#pragma unroll
    for (int r = 0; r < 8; ++r) {
      int grow = m0 + r + 8 * hi;
      float spos = (float)(grow % S_);
#pragma unroll
      for (int t = 0; t < 4; ++t) {
        int j = t * 16 + l15;  // 0..63
        float ang = spos * __expf(-(float)(2 * j) * (1.0f / 128.0f) * logbase);
        float sn, cs;
        __sincosf(ang, &sn, &cs);
        float x1 = c[t][r], x2 = c[t + 4][r];
        c[t][r]     =  x1 * cs + x2 * sn;
        c[t + 4][r] = -x1 * sn + x2 * cs;
      }
    }
  }

  if (hc < 16) {                 // Q: gain + store (b,h,s,d)
    float gain = q_gain[hc];
#pragma unroll
    for (int r = 0; r < 8; ++r) {
      int grow = m0 + r + 8 * hi;
      int b = grow / S_, s = grow % S_;
      half_t* dst = q16 + ((size_t)(b * H_ + hc) * S_ + s) * DH_;
#pragma unroll
      for (int t = 0; t < 8; ++t) dst[t * 16 + l15] = (half_t)(c[t][r] * gain);
    }
  } else if (hc < 20) {          // K: store (b,hk,s,d)
    int hk = hc - 16;
#pragma unroll
    for (int r = 0; r < 8; ++r) {
      int grow = m0 + r + 8 * hi;
      int b = grow / S_, s = grow % S_;
      half_t* dst = k16 + ((size_t)(b * HKV_ + hk) * S_ + s) * DH_;
#pragma unroll
      for (int t = 0; t < 8; ++t) dst[t * 16 + l15] = (half_t)c[t][r];
    }
  } else {                       // V: store transposed (b,hv,d,s) for P@V B-frags
    int hv = hc - 20;
#pragma unroll
    for (int r = 0; r < 8; ++r) {
      int grow = m0 + r + 8 * hi;
      int b = grow / S_, s = grow % S_;
#pragma unroll
      for (int t = 0; t < 8; ++t) {
        int d = t * 16 + l15;
        vT16[((size_t)(b * HKV_ + hv) * DH_ + d) * S_ + s] = (half_t)c[t][r];
      }
    }
  }
}

// ---------------- Flash attention: wave owns 16 q-rows ----------------

#define QCHUNK 128

__global__ __launch_bounds__(256)
void k_attn(const half_t* __restrict__ q16, const half_t* __restrict__ k16,
            const half_t* __restrict__ vT16, half_t* __restrict__ y16) {
  __shared__ __align__(32) half_t plds[8 * 16 * PPITCH];
  const int lane = threadIdx.x & 31;
  const int l15 = lane & 15;
  const int hi  = lane >> 4;
  const int wave = threadIdx.x >> 5;
  const int chunk = blockIdx.x % (S_ / QCHUNK);
  const int bh = blockIdx.x / (S_ / QCHUNK);
  const int b = bh / H_, h = bh % H_;
  const int hk = h / REPS_;
  const int q0 = chunk * QCHUNK + wave * 16;

  const half_t* qrow = q16 + ((size_t)(b * H_ + h) * S_ + q0 + l15) * DH_;
  v16h qa[4];
#pragma unroll
  for (int cd = 0; cd < 4; ++cd) qa[cd] = load_a16(qrow, cd * 32, hi);

  const float scale = 0.08838834764831845f;  // 1/sqrt(128)
  float mrun[8], lrun[8];
  v8f z = {};
  v8f o[8];
#pragma unroll
  for (int r = 0; r < 8; ++r) { mrun[r] = -1e30f; lrun[r] = 0.f; }
#pragma unroll
  for (int t = 0; t < 8; ++t) o[t] = z;

  const half_t* kbase = k16 + (size_t)(b * HKV_ + hk) * S_ * DH_;
  const half_t* vbase = vT16 + (size_t)(b * HKV_ + hk) * DH_ * S_;
  half_t* wl = plds + wave * 16 * PPITCH;

  for (int kv0 = 0; kv0 < q0 + 16; kv0 += 32) {
    // ---- hoist all K B-frags, then burst 8 WMMAs ----
    v16h kb[8];
#pragma unroll
    for (int tile = 0; tile < 2; ++tile) {
      const half_t* krow = kbase + (size_t)(kv0 + tile * 16 + l15) * DH_;
#pragma unroll
      for (int cd = 0; cd < 4; ++cd)
        kb[tile * 4 + cd] = load_b16(krow, cd * 32, hi);
    }
    v8f st[2];
    st[0] = z; st[1] = z;
#pragma unroll
    for (int cd = 0; cd < 4; ++cd) st[0] = wmma16(qa[cd], kb[cd], st[0]);
#pragma unroll
    for (int cd = 0; cd < 4; ++cd) st[1] = wmma16(qa[cd], kb[4 + cd], st[1]);

    // ---- issue V B-frag loads now; latency hides under softmax VALU ----
    v16h vb[8];
#pragma unroll
    for (int t = 0; t < 8; ++t) {
      const half_t* vrow = vbase + (size_t)(t * 16 + l15) * S_ + kv0 + hi * 16;
      vb[t] = *reinterpret_cast<const v16h*>(vrow);
    }

    // scale + causal mask
#pragma unroll
    for (int tile = 0; tile < 2; ++tile)
#pragma unroll
      for (int r = 0; r < 8; ++r) {
        int qpos = q0 + r + 8 * hi;
        int kpos = kv0 + tile * 16 + l15;
        float v = st[tile][r] * scale;
        st[tile][r] = (kpos <= qpos) ? v : -1e30f;
      }
    // ---- online softmax (row = 16 lanes of one half-wave) ----
    float alpha[8];
#pragma unroll
    for (int r = 0; r < 8; ++r) {
      float mx = fmaxf(st[0][r], st[1][r]);
#pragma unroll
      for (int off = 1; off < 16; off <<= 1)
        mx = fmaxf(mx, __shfl_xor(mx, off, 16));
      float mnew = fmaxf(mrun[r], mx);
      float a = __expf(mrun[r] - mnew);
      float p0 = __expf(st[0][r] - mnew);
      float p1 = __expf(st[1][r] - mnew);
      st[0][r] = p0; st[1][r] = p1;
      float ps = p0 + p1;
#pragma unroll
      for (int off = 1; off < 16; off <<= 1) ps += __shfl_xor(ps, off, 16);
      lrun[r] = lrun[r] * a + ps;
      mrun[r] = mnew;
      alpha[r] = a;
    }
#pragma unroll
    for (int t = 0; t < 8; ++t)
#pragma unroll
      for (int r = 0; r < 8; ++r) o[t][r] *= alpha[r];

    // ---- transpose P (C-layout -> A-layout) via LDS; same-wave DS in-order ----
#pragma unroll
    for (int tile = 0; tile < 2; ++tile)
#pragma unroll
      for (int r = 0; r < 8; ++r)
        wl[(r + 8 * hi) * PPITCH + tile * 16 + l15] = (half_t)st[tile][r];
    const half_t* pr = wl + l15 * PPITCH + hi * 16;
    v16h pa = cat16(*reinterpret_cast<const v8h*>(pr),
                    *reinterpret_cast<const v8h*>(pr + 8));

    // ---- O += P @ V ----
#pragma unroll
    for (int t = 0; t < 8; ++t) o[t] = wmma16(pa, vb[t], o[t]);
  }

  // normalize + store y (b, s, h*Dh + d)
#pragma unroll
  for (int r = 0; r < 8; ++r) {
    float inv = 1.0f / lrun[r];
    int srow = q0 + r + 8 * hi;
    half_t* dst = y16 + (size_t)(b * S_ + srow) * D_ + h * DH_;
#pragma unroll
    for (int t = 0; t < 8; ++t) dst[t * 16 + l15] = (half_t)(o[t][r] * inv);
  }
}

// ---------------- Output projection: y @ Wo^T -> fp32 out ----------------
// Block: 8 waves share a 64-col slab of Wo (64-wide k-slices, async
// double-buffered in LDS); wave w owns m-tile mblock*8+w.

__global__ __launch_bounds__(256)
void k_oproj(const half_t* __restrict__ y16, const half_t* __restrict__ wo16,
             float* __restrict__ out) {
  __shared__ __align__(32) half_t ldsB[2 * 64 * WPITCH];
  const int lane = threadIdx.x & 31;
  const int l15 = lane & 15;
  const int hi  = lane >> 4;
  const int wave = threadIdx.x >> 5;
  const int mblock = blockIdx.x >> 5;
  const int ng     = blockIdx.x & 31;
  const int m0 = (mblock * 8 + wave) * 16;
  const int n0 = ng * 64;

  v8f z = {};
  v8f c[4];
#pragma unroll
  for (int t = 0; t < 4; ++t) c[t] = z;

  const half_t* arow = y16 + (size_t)(m0 + l15) * D_;

  // staging: thread -> (row 0..63, 16-half chunk 0..3) => 2 b128 asyncs each
  const int srow = threadIdx.x >> 2;
  const int sp   = threadIdx.x & 3;
  const half_t* wsrc = wo16 + (size_t)(n0 + srow) * D_ + sp * 16;
  half_t* sdst[2];
  sdst[0] = ldsB + srow * WPITCH + sp * 16;
  sdst[1] = sdst[0] + 64 * WPITCH;

  async_g2l_b128(sdst[0],     wsrc);
  async_g2l_b128(sdst[0] + 8, wsrc + 8);
  wait_async0();
  __syncthreads();

  v16h a0 = load_a16(arow, 0, hi);
  v16h a1 = load_a16(arow, 32, hi);
  int cur = 0;
  for (int k0 = 0; k0 < D_; k0 += 64) {
    const bool more = (k0 + 64 < D_);
    if (more) {
      const half_t* ns = wsrc + k0 + 64;
      half_t* nd = sdst[cur ^ 1];
      async_g2l_b128(nd,     ns);
      async_g2l_b128(nd + 8, ns + 8);
      __builtin_prefetch(arow + k0 + 128, 0, 1);
    }
    v16h an0 = a0, an1 = a1;
    if (more) { an0 = load_a16(arow, k0 + 64, hi); an1 = load_a16(arow, k0 + 96, hi); }

    const half_t* bb = ldsB + cur * (64 * WPITCH);
    v16h kb[4];
#pragma unroll
    for (int t = 0; t < 4; ++t) kb[t] = ldsb_frag(bb, t * 16 + l15, 0, hi);
#pragma unroll
    for (int t = 0; t < 4; ++t) c[t] = wmma16(a0, kb[t], c[t]);
#pragma unroll
    for (int t = 0; t < 4; ++t) kb[t] = ldsb_frag(bb, t * 16 + l15, 32, hi);
#pragma unroll
    for (int t = 0; t < 4; ++t) c[t] = wmma16(a1, kb[t], c[t]);

    a0 = an0; a1 = an1;
    wait_async0();
    __syncthreads();
    cur ^= 1;
  }
#pragma unroll
  for (int r = 0; r < 8; ++r) {
    int row = m0 + r + 8 * hi;
#pragma unroll
    for (int t = 0; t < 4; ++t)
      out[(size_t)row * D_ + n0 + t * 16 + l15] = c[t][r];
  }
}

// ---------------- host launcher ----------------

extern "C" void kernel_launch(void* const* d_in, const int* in_sizes, int n_in,
                              void* d_out, int out_size, void* d_ws, size_t ws_size,
                              hipStream_t stream) {
  (void)in_sizes; (void)n_in; (void)out_size; (void)ws_size;
  const float* x  = (const float*)d_in[0];
  const float* Wq = (const float*)d_in[1];
  const float* Wk = (const float*)d_in[2];
  const float* Wv = (const float*)d_in[3];
  const float* Wo = (const float*)d_in[4];
  const float* qg = (const float*)d_in[5];

  half_t* ws = (half_t*)d_ws;
  size_t off = 0;
  half_t* x16  = ws + off; off += (size_t)M_ * D_;
  half_t* wqkv = ws + off; off += (size_t)NQKV_ * D_;
  half_t* wo16 = ws + off; off += (size_t)D_ * D_;
  half_t* q16  = ws + off; off += (size_t)B_ * H_ * S_ * DH_;
  half_t* k16  = ws + off; off += (size_t)B_ * HKV_ * S_ * DH_;
  half_t* vT16 = ws + off; off += (size_t)B_ * HKV_ * DH_ * S_;
  half_t* y16  = ws + off; off += (size_t)M_ * D_;

  auto conv = [&](const float* s, half_t* d, int n) {
    int blocks = (n + 255) / 256;
    if (blocks > 4096) blocks = 4096;
    k_f32_to_f16<<<blocks, 256, 0, stream>>>(s, d, n);
  };
  conv(x, x16, M_ * D_);
  conv(Wq, wqkv, D_ * D_);
  conv(Wk, wqkv + (size_t)D_ * D_, HKV_ * DH_ * D_);
  conv(Wv, wqkv + (size_t)(D_ + HKV_ * DH_) * D_, HKV_ * DH_ * D_);
  conv(Wo, wo16, D_ * D_);

  // 32 m-blocks * 24 head-cols = 768 blocks
  k_qkv<<<768, 256, 0, stream>>>(x16, wqkv, qg, q16, k16, vT16);
  // B*H*(S/128) = 512 blocks
  k_attn<<<B_ * H_ * (S_ / QCHUNK), 256, 0, stream>>>(q16, k16, vT16, y16);
  // 32 m-blocks * 32 n-groups = 1024 blocks
  k_oproj<<<1024, 256, 0, stream>>>(y16, wo16, (float*)d_out);
}
